// Attention_5798205850042
// MI455X (gfx1250) — matrix operand order repaired and checked by reading, not compiled, not run on hardware
//
#include <hip/hip_runtime.h>

// ---------------------------------------------------------------------------
// CDNA5 (gfx1250) wave32 WMMA attention:
//   1) QKV GEMM 256x128 tile, double-buffered, 16 wmma/chunk/wave
//   2) per-head q/k LayerNorm + repack to [B,H,N,D] f16 (ds_swizzle reductions)
//   3) flash attention, 64-key steps: 18 wmma/step, row-sum via P@ones wmma,
//      row-max via ds_swizzle butterflies only
//   4) proj GEMM (same kernel as 1, f16 A, f32 out)
// ---------------------------------------------------------------------------

typedef __attribute__((ext_vector_type(16))) _Float16 v16h;
typedef __attribute__((ext_vector_type(8)))  _Float16 h8;
typedef __attribute__((ext_vector_type(4)))  _Float16 h4;
typedef __attribute__((ext_vector_type(8)))  float    v8f;

static __device__ __forceinline__ v16h frag16(h8 lo, h8 hi) {
  return __builtin_shufflevector(lo, hi, 0,1,2,3,4,5,6,7,8,9,10,11,12,13,14,15);
}

static __device__ __forceinline__ v8f wmma_f16(v16h a, v16h b, v8f c) {
  return __builtin_amdgcn_wmma_f32_16x16x32_f16(false, a, false, b, (short)0, c,
                                                false, false);
}

// xor-butterfly via DS_SWIZZLE_B32 (group-of-32 mode: and=0x1f, or=0, xor=MASK)
template <int MASK>
static __device__ __forceinline__ float swz_xor(float v) {
  int r = __builtin_amdgcn_ds_swizzle(__float_as_int(v), (MASK << 10) | 0x1F);
  return __int_as_float(r);
}

// Load a 32-element row slice (f32 or f16) from global, as 4x h8 (f16).
template <typename AT>
static __device__ __forceinline__ void load_row32(const AT* p, h8 out[4]) {
  if constexpr (sizeof(AT) == 2) {
    const h8* v = (const h8*)p;
#pragma unroll
    for (int i = 0; i < 4; ++i) out[i] = v[i];
  } else {
    const float4* v = (const float4*)p;
#pragma unroll
    for (int i = 0; i < 4; ++i) {
      float4 f0 = v[2 * i], f1 = v[2 * i + 1];
      h8 h;
      h[0] = (_Float16)f0.x; h[1] = (_Float16)f0.y;
      h[2] = (_Float16)f0.z; h[3] = (_Float16)f0.w;
      h[4] = (_Float16)f1.x; h[5] = (_Float16)f1.y;
      h[6] = (_Float16)f1.z; h[7] = (_Float16)f1.w;
      out[i] = h;
    }
  }
}

static __device__ __forceinline__ void load_row16(const float* p, h8 out[2]) {
  const float4* v = (const float4*)p;
#pragma unroll
  for (int i = 0; i < 2; ++i) {
    float4 f0 = v[2 * i], f1 = v[2 * i + 1];
    h8 h;
    h[0] = (_Float16)f0.x; h[1] = (_Float16)f0.y;
    h[2] = (_Float16)f0.z; h[3] = (_Float16)f0.w;
    h[4] = (_Float16)f1.x; h[5] = (_Float16)f1.y;
    h[6] = (_Float16)f1.z; h[7] = (_Float16)f1.w;
    out[i] = h;
  }
}

// ---------------------------------------------------------------------------
// Kernel 1/4: Y[M,N] = A[M,K] @ W[N,K]^T + bias.
// Block tile 256x128, K-chunk 32, double-buffered staging.
// 8 waves (4m x 2n), each wave 64x64 = 4x4 WMMA accumulators.
// ---------------------------------------------------------------------------
template <typename AT, typename OT>
__global__ __launch_bounds__(256) void gemm_wmma_kernel(
    const AT* __restrict__ A, const float* __restrict__ W,
    const float* __restrict__ bias, OT* __restrict__ Y,
    int M, int N, int K) {
  __shared__ _Float16 sA[256][32];   // [m][k]
  __shared__ _Float16 sB[128][32];   // [n][k]

  const int t    = threadIdx.x;
  const int wave = t >> 5;
  const int lane = t & 31;
  const int lrow = lane & 15;
  const int lh   = lane >> 4;
  const int kh   = lh * 8;                // A-fragment K phase
  const int wm   = wave >> 1;             // 0..3
  const int wn   = wave & 1;              // 0..1
  const int bm   = blockIdx.y * 256;
  const int bn   = blockIdx.x * 128;

  const AT*    aRow = A + (size_t)(bm + t) * K;
  const float* bRow = W + (size_t)(bn + (t >> 1)) * K;
  const int    bcol = (t & 1) * 16;

  v8f zero = {};
  v8f c[4][4];
#pragma unroll
  for (int i = 0; i < 4; ++i)
#pragma unroll
    for (int j = 0; j < 4; ++j) c[i][j] = zero;

  h8 ra[4], rb[2];
  load_row32<AT>(aRow, ra);
  load_row16(bRow + bcol, rb);

  for (int kt = 0; kt < K; kt += 32) {
    {  // commit staged chunk to LDS
      h8* d = (h8*)&sA[t][0];
      d[0] = ra[0]; d[1] = ra[1]; d[2] = ra[2]; d[3] = ra[3];
      h8* e = (h8*)&sB[t >> 1][bcol];
      e[0] = rb[0]; e[1] = rb[1];
    }
    __syncthreads();
    if (kt + 32 < K) {  // prefetch next chunk while WMMAs run
      load_row32<AT>(aRow + kt + 32, ra);
      load_row16(bRow + bcol + kt + 32, rb);
    }

    v16h af[4], bf[4];
#pragma unroll
    for (int i = 0; i < 4; ++i) {
      const int mr = wm * 64 + i * 16 + lrow;
      af[i] = frag16(*(const h8*)&sA[mr][kh], *(const h8*)&sA[mr][kh + 16]);
    }
#pragma unroll
    for (int j = 0; j < 4; ++j) {
      const int nr = wn * 64 + j * 16 + lrow;
      bf[j] = frag16(*(const h8*)&sB[nr][lh * 8],
                     *(const h8*)&sB[nr][lh * 8 + 16]);
    }
#pragma unroll
    for (int i = 0; i < 4; ++i)
#pragma unroll
      for (int j = 0; j < 4; ++j) c[i][j] = wmma_f16(af[i], bf[j], c[i][j]);
    __syncthreads();
  }

#pragma unroll
  for (int j = 0; j < 4; ++j) {
    const int col = bn + wn * 64 + j * 16 + lrow;
    const float bv = bias[col];
#pragma unroll
    for (int i = 0; i < 4; ++i)
#pragma unroll
      for (int r = 0; r < 8; ++r) {
        const int row = bm + wm * 64 + i * 16 + r + 8 * lh;
        Y[(size_t)row * N + col] = (OT)(c[i][j][r] + bv);
      }
  }
}

// ---------------------------------------------------------------------------
// Kernel 2: per-head LayerNorm of q,k (+1/sqrt(D) on q), repack to [B,H,N,64].
// ---------------------------------------------------------------------------
__global__ __launch_bounds__(256) void ln_pack_kernel(
    const _Float16* __restrict__ qkv,  // [B*N][which][h][d]
    const float* __restrict__ qw, const float* __restrict__ qb,
    const float* __restrict__ kw, const float* __restrict__ kb,
    _Float16* __restrict__ Qn, _Float16* __restrict__ Kn,
    _Float16* __restrict__ Vh) {
  const int t   = threadIdx.x;
  const int l16 = t & 15;
  const int rid = blockIdx.x * 16 + (t >> 4);
  const int h   = rid & 15;
  const int tmp = rid >> 4;
  const int which = tmp % 3;
  const int bn    = tmp / 3;   // b*4096 + n

  const _Float16* src =
      qkv + (size_t)bn * 3072 + which * 1024 + h * 64 + l16 * 4;
  h4 xv = *(const h4*)src;
  float x[4];
#pragma unroll
  for (int i = 0; i < 4; ++i) x[i] = (float)xv[i];

  float s = x[0] + x[1] + x[2] + x[3];
  float q = x[0] * x[0] + x[1] * x[1] + x[2] * x[2] + x[3] * x[3];
  s += swz_xor<1>(s); q += swz_xor<1>(q);
  s += swz_xor<2>(s); q += swz_xor<2>(q);
  s += swz_xor<4>(s); q += swz_xor<4>(q);
  s += swz_xor<8>(s); q += swz_xor<8>(q);

  const float mu   = s * (1.0f / 64.0f);
  const float var  = q * (1.0f / 64.0f) - mu * mu;
  const float rstd = rsqrtf(var + 1e-5f);

  h4 out;
  if (which == 2) {
    out = xv;
  } else {
    const float* w  = (which == 0) ? qw : kw;
    const float* bb = (which == 0) ? qb : kb;
    const float sc  = (which == 0) ? 0.125f : 1.0f;
#pragma unroll
    for (int i = 0; i < 4; ++i) {
      const int d = l16 * 4 + i;
      out[i] = (_Float16)(((x[i] - mu) * rstd * w[d] + bb[d]) * sc);
    }
  }

  _Float16* dstbase = (which == 0) ? Qn : (which == 1) ? Kn : Vh;
  const int b_ = bn >> 12;
  const int n  = bn & 4095;
  *(h4*)(dstbase + (((size_t)(b_ * 16 + h)) * 4096 + n) * 64 + l16 * 4) = out;
}

// ---------------------------------------------------------------------------
// Kernel 3: flash attention.  grid = (N/128, B*H), 8 waves x 16 query rows.
// 64-key steps: 8 WMMA (S) + softmax (max-only butterflies) + 10 WMMA (O, l).
// ---------------------------------------------------------------------------
__global__ __launch_bounds__(256) void flash_attn_kernel(
    const _Float16* __restrict__ Q,   // [B*H, N, 64] (scale folded in)
    const _Float16* __restrict__ Kk,  // [B*H, N, 64]
    const _Float16* __restrict__ V,   // [B*H, N, 64]
    _Float16* __restrict__ O,         // [B*N, 1024]
    int Nn) {
  __shared__ _Float16 sK[64][64];      // [key][d]   (B-frag layout for S)
  __shared__ _Float16 sV[64][64];      // [d][key]   (B-frag layout for O)
  __shared__ _Float16 sP[8][16][64];   // per-wave P round-trip

  const int t    = threadIdx.x;
  const int wave = t >> 5;
  const int lane = t & 31;
  const int lrow = lane & 15;
  const int lh   = lane >> 4;
  const int kh   = lh * 8;
  const int bh   = blockIdx.y;
  const int b_   = bh >> 4;
  const int h    = bh & 15;
  const int q0   = blockIdx.x * 128 + wave * 16;

  // Q fragments, register-resident
  const _Float16* qbase = Q + ((size_t)bh * Nn + q0 + lrow) * 64;
  v16h qf[2];
#pragma unroll
  for (int dc = 0; dc < 2; ++dc)
    qf[dc] = frag16(*(const h8*)(qbase + dc * 32 + kh),
                    *(const h8*)(qbase + dc * 32 + kh + 16));

  // constant ones B-fragment: l = P @ ones (row-sum as a WMMA column tile)
  v16h onesf;
#pragma unroll
  for (int i = 0; i < 16; ++i) onesf[i] = (_Float16)1.0f;

  v8f zero = {};
  v8f o[4], o4;
#pragma unroll
  for (int i = 0; i < 4; ++i) o[i] = zero;
  o4 = zero;
  float mrow[8];
#pragma unroll
  for (int r = 0; r < 8; ++r) mrow[r] = -1e30f;

  const int krow = t >> 2;          // 0..63 key index for staging
  const int kd0  = (t & 3) * 16;    // d offset

  for (int kb = 0; kb < Nn; kb += 64) {
    const size_t gs = ((size_t)bh * Nn + kb + krow) * 64 + kd0;
    h8 kv0 = *(const h8*)(Kk + gs);
    h8 kv1 = *(const h8*)(Kk + gs + 8);
    *(h8*)&sK[krow][kd0]     = kv0;
    *(h8*)&sK[krow][kd0 + 8] = kv1;
    h8 vv0 = *(const h8*)(V + gs);
    h8 vv1 = *(const h8*)(V + gs + 8);
#pragma unroll
    for (int i = 0; i < 8; ++i) {
      sV[kd0 + i][krow]     = vv0[i];
      sV[kd0 + 8 + i][krow] = vv1[i];
    }
    __syncthreads();

    // S = Q @ K^T : 4 key subtiles x 2 d-chunks
    v8f s[4];
#pragma unroll
    for (int ns = 0; ns < 4; ++ns) {
      s[ns] = zero;
#pragma unroll
      for (int dc = 0; dc < 2; ++dc) {
        h8 lo = *(const h8*)&sK[ns * 16 + lrow][dc * 32 + lh * 8];
        h8 hi = *(const h8*)&sK[ns * 16 + lrow][dc * 32 + lh * 8 + 16];
        s[ns] = wmma_f16(qf[dc], frag16(lo, hi), s[ns]);
      }
    }

    // online softmax: row-max butterflies only (row-sum handled by ones-WMMA)
#pragma unroll
    for (int r = 0; r < 8; ++r) {
      float rmax = fmaxf(fmaxf(s[0][r], s[1][r]), fmaxf(s[2][r], s[3][r]));
      rmax = fmaxf(rmax, swz_xor<8>(rmax));
      rmax = fmaxf(rmax, swz_xor<4>(rmax));
      rmax = fmaxf(rmax, swz_xor<2>(rmax));
      rmax = fmaxf(rmax, swz_xor<1>(rmax));
      const float mnew  = fmaxf(mrow[r], rmax);
      const float alpha = __expf(mrow[r] - mnew);
      mrow[r] = mnew;
#pragma unroll
      for (int tt = 0; tt < 4; ++tt) o[tt][r] *= alpha;
      o4[r] *= alpha;
#pragma unroll
      for (int ns = 0; ns < 4; ++ns)
        sP[wave][r + 8 * lh][ns * 16 + lrow] =
            (_Float16)__expf(s[ns][r] - mnew);
    }
    __builtin_amdgcn_wave_barrier();  // D-layout stores before A-frag loads

    // O += P @ V ; o4 += P @ ones (running softmax denominator)
#pragma unroll
    for (int pc = 0; pc < 2; ++pc) {
      h8 plo = *(const h8*)&sP[wave][lrow][pc * 32 + kh];
      h8 phi = *(const h8*)&sP[wave][lrow][pc * 32 + kh + 16];
      v16h pf = frag16(plo, phi);
#pragma unroll
      for (int tt = 0; tt < 4; ++tt) {
        h8 lo = *(const h8*)&sV[tt * 16 + lrow][pc * 32 + lh * 8];
        h8 hi = *(const h8*)&sV[tt * 16 + lrow][pc * 32 + lh * 8 + 16];
        o[tt] = wmma_f16(pf, frag16(lo, hi), o[tt]);
      }
      o4 = wmma_f16(pf, onesf, o4);
    }
    __syncthreads();
  }

  // epilogue: O /= l, write [b][n][h*64+d] f16
#pragma unroll
  for (int r = 0; r < 8; ++r) {
    const float inv = 1.0f / o4[r];
    const int nrow  = q0 + r + 8 * lh;
#pragma unroll
    for (int tt = 0; tt < 4; ++tt) {
      const int col = h * 64 + tt * 16 + lrow;
      O[((size_t)b_ * Nn + nrow) * 1024 + col] = (_Float16)(o[tt][r] * inv);
    }
  }
}

// ---------------------------------------------------------------------------
extern "C" void kernel_launch(void* const* d_in, const int* in_sizes, int n_in,
                              void* d_out, int out_size, void* d_ws,
                              size_t ws_size, hipStream_t stream) {
  (void)in_sizes; (void)n_in; (void)out_size; (void)ws_size;
  const float* x      = (const float*)d_in[0];
  const float* qkv_w  = (const float*)d_in[1];
  const float* qkv_b  = (const float*)d_in[2];
  const float* qnw    = (const float*)d_in[3];
  const float* qnb    = (const float*)d_in[4];
  const float* knw    = (const float*)d_in[5];
  const float* knb    = (const float*)d_in[6];
  const float* proj_w = (const float*)d_in[7];
  const float* proj_b = (const float*)d_in[8];

  const int M  = 8192;   // B*N
  const int C  = 1024;
  const int C3 = 3072;
  const int Nn = 4096;

  char* ws = (char*)d_ws;
  _Float16* qkvh = (_Float16*)ws;                          // 48 MiB
  size_t off = (size_t)M * C3 * 2;
  _Float16* Qn   = (_Float16*)(ws + off); off += (size_t)M * C * 2;
  _Float16* Kn   = (_Float16*)(ws + off); off += (size_t)M * C * 2;
  _Float16* Vh   = (_Float16*)(ws + off); off += (size_t)M * C * 2;
  _Float16* attn = (_Float16*)(ws + off);

  dim3 blk(256);

  gemm_wmma_kernel<float, _Float16>
      <<<dim3(C3 / 128, M / 256), blk, 0, stream>>>(x, qkv_w, qkv_b, qkvh,
                                                    M, C3, C);

  ln_pack_kernel<<<dim3(M * 3), blk, 0, stream>>>(qkvh, qnw, qnb, knw, knb,
                                                  Qn, Kn, Vh);

  flash_attn_kernel<<<dim3(Nn / 128, 32), blk, 0, stream>>>(Qn, Kn, Vh, attn,
                                                            Nn);

  gemm_wmma_kernel<_Float16, float>
      <<<dim3(C / 128, M / 256), blk, 0, stream>>>(attn, proj_w, proj_b,
                                                   (float*)d_out, M, C, C);
}
